// Transformer_18408229831105
// MI455X (gfx1250) — matrix operand order repaired and checked by reading, not compiled, not run on hardware
//
#include <hip/hip_runtime.h>
#include <hip/hip_bf16.h>
#include <cstddef>

// ---------------------------------------------------------------------------
// Model dims (compile-time constants from the reference)
// ---------------------------------------------------------------------------
#define BD   2
#define SD   512
#define DD   768
#define VD   32000
#define NLD  2
#define NED  8
#define NTD  2
#define TDD  32
#define ATTRD 64
#define INTERD 1536
#define MEMD 256
#define NHD  4
#define HDD  192          // DD / NHD
#define TOK  (BD*SD)      // 1024 tokens

typedef __attribute__((ext_vector_type(16))) _Float16 v16h;
typedef __attribute__((ext_vector_type(8)))  float    v8f;
typedef int vi4 __attribute__((vector_size(16)));   // 128-bit payload type of the async builtin

#define AS1 __attribute__((address_space(1)))
#define AS3 __attribute__((address_space(3)))

#ifndef __has_builtin
#define __has_builtin(x) 0
#endif
#if __has_builtin(__builtin_amdgcn_global_load_async_to_lds_b128)
#define HAVE_ASYNC 1
#else
#define HAVE_ASYNC 0
#endif

__device__ __forceinline__ void cvt4(v16h& v, int base, const float4 f)
{
    v[base + 0] = (_Float16)f.x;
    v[base + 1] = (_Float16)f.y;
    v[base + 2] = (_Float16)f.z;
    v[base + 3] = (_Float16)f.w;
}

// ---------------------------------------------------------------------------
// WMMA GEMM with async global->LDS double buffering.
//   C[m,n] = act( sum_k A[m,k]*W[n,k] + bias[n] )
// A: M x K (row stride lda), W: N x K (row stride ldw), C: M x N (stride ldc)
// act: 0 = none, 1 = relu, 2 = sigmoid
// grid = (ceil(N/128), ceil(M/64)), block = 256 (8 waves: 4 M-groups x 2 N-groups)
// Each k-step (32): A tile 64x32 f32 (8KB) + B tile 128x32 f32 (16KB) staged in
// LDS via GLOBAL_LOAD_ASYNC_TO_LDS_B128 (6 per thread), double-buffered and
// overlapped with the 4 WMMAs/wave of the previous step. LDS rows padded to 36
// floats -> conflict-free ds_load_b128 fragment reads. All fragments are built
// into distinct registers before the 4 back-to-back WMMAs (no WAR hazard NOPs,
// all 20 ds_loads in flight at once). Ragged-N blocks (N=2 / N=32 GEMMs) take
// a uniform fallback that zero-fills LDS; compute loop stays branch-free.
// ---------------------------------------------------------------------------
#define LPAD 36

__global__ void __launch_bounds__(256)
k_gemm_wmma(const float* __restrict__ A, int lda,
            const float* __restrict__ W, int ldw,
            const float* __restrict__ bias,
            float* __restrict__ C, int ldc,
            int M, int N, int K, int act)
{
    __shared__ __align__(16) float As[2][64][LPAD];
    __shared__ __align__(16) float Bs[2][128][LPAD];

    const int tid  = threadIdx.x;
    const int lane = tid & 31;
    const int wave = tid >> 5;
    const int wm   = wave & 3;
    const int wn   = wave >> 2;
    const int mblk = blockIdx.y * 64;
    const int nblk = blockIdx.x * 128;

    const int hlf  = lane >> 4;     // lane half: 0 or 1
    const int lidx = lane & 15;

    const bool fast = (mblk + 64 <= M) && (nblk + 128 <= N);
    const int  nk   = K >> 5;       // K is always a multiple of 32 here

    // tile-fill decomposition: each thread owns one 16B chunk per 32 rows
    const int frow = tid >> 3;         // 0..31
    const int fcol = (tid & 7) * 4;    // 0,4,...,28

    auto fill = [&](int s, int kb) {
        if (fast) {
#if HAVE_ASYNC
            #pragma unroll
            for (int q = 0; q < 2; ++q) {
                const int row = frow + q * 32;
                const float* gp = A + (size_t)(mblk + row) * lda + kb + fcol;
                __builtin_amdgcn_global_load_async_to_lds_b128(
                    (AS1 vi4*)gp, (AS3 vi4*)&As[s][row][fcol], 0, 0);
            }
            #pragma unroll
            for (int q = 0; q < 4; ++q) {
                const int row = frow + q * 32;
                const float* gp = W + (size_t)(nblk + row) * ldw + kb + fcol;
                __builtin_amdgcn_global_load_async_to_lds_b128(
                    (AS1 vi4*)gp, (AS3 vi4*)&Bs[s][row][fcol], 0, 0);
            }
#else
            #pragma unroll
            for (int q = 0; q < 2; ++q) {
                const int row = frow + q * 32;
                *(float4*)&As[s][row][fcol] =
                    *(const float4*)(A + (size_t)(mblk + row) * lda + kb + fcol);
            }
            #pragma unroll
            for (int q = 0; q < 4; ++q) {
                const int row = frow + q * 32;
                *(float4*)&Bs[s][row][fcol] =
                    *(const float4*)(W + (size_t)(nblk + row) * ldw + kb + fcol);
            }
#endif
        } else {
            #pragma unroll
            for (int q = 0; q < 2; ++q) {
                const int row = frow + q * 32;
                float4 v = make_float4(0.f, 0.f, 0.f, 0.f);
                if (mblk + row < M)
                    v = *(const float4*)(A + (size_t)(mblk + row) * lda + kb + fcol);
                *(float4*)&As[s][row][fcol] = v;
            }
            #pragma unroll
            for (int q = 0; q < 4; ++q) {
                const int row = frow + q * 32;
                float4 v = make_float4(0.f, 0.f, 0.f, 0.f);
                if (nblk + row < N)
                    v = *(const float4*)(W + (size_t)(nblk + row) * ldw + kb + fcol);
                *(float4*)&Bs[s][row][fcol] = v;
            }
        }
    };

    v8f acc[4] = {};
    fill(0, 0);

    for (int i = 0; i < nk; ++i) {
        if (i + 1 < nk) fill((i + 1) & 1, (i + 1) << 5);
#if HAVE_ASYNC
        if (fast) {
            if (i + 1 < nk) asm volatile("s_wait_asynccnt 6" ::: "memory");
            else            asm volatile("s_wait_asynccnt 0" ::: "memory");
        }
#endif
        __syncthreads();
        const int s = i & 1;

        // ---- stage all fragment data from LDS first (20 ds_load_b128 in flight)
        float4 at[4];
        {
            const float* ap = &As[s][wm * 16 + lidx][hlf * 16];
            at[0] = *(const float4*)(ap + 0);
            at[1] = *(const float4*)(ap + 4);
            at[2] = *(const float4*)(ap + 8);
            at[3] = *(const float4*)(ap + 12);
        }
        float4 bt[4][4];
        #pragma unroll
        for (int j = 0; j < 4; ++j) {
            const float* bp = &Bs[s][wn * 64 + j * 16 + lidx][hlf * 16];
            bt[j][0] = *(const float4*)(bp + 0);
            bt[j][1] = *(const float4*)(bp + 4);
            bt[j][2] = *(const float4*)(bp + 8);
            bt[j][3] = *(const float4*)(bp + 12);
        }

        // ---- convert to f16 fragments
        // A (16x32): lanes 0-15 K=0..15, lanes 16-31 K=16..31
        v16h a;
        cvt4(a, 0, at[0]); cvt4(a, 4, at[1]); cvt4(a, 8, at[2]); cvt4(a, 12, at[3]);
        // B (32x16): lanes 0-15 K=0..15, lanes 16-31 K=16..31
        v16h bfr[4];
        #pragma unroll
        for (int j = 0; j < 4; ++j) {
            cvt4(bfr[j], 0,  bt[j][0]);
            cvt4(bfr[j], 4,  bt[j][1]);
            cvt4(bfr[j], 8,  bt[j][2]);
            cvt4(bfr[j], 12, bt[j][3]);
        }

        // ---- 4 back-to-back WMMAs, no VALU writes to their sources in between
        #pragma unroll
        for (int j = 0; j < 4; ++j)
            acc[j] = __builtin_amdgcn_wmma_f32_16x16x32_f16(
                        false, a, false, bfr[j], (short)0, acc[j], false, false);

        __syncthreads();   // protect buffer (i&1) before it is refilled at i+2
    }

    // ---- store: C/D layout -> VGPR r: lanes 0-15 M=r, lanes 16-31 M=r+8; N=lane&15
    const int m0 = mblk + wm * 16;
    const int n0 = nblk + wn * 64;
    #pragma unroll
    for (int j = 0; j < 4; ++j) {
        const int n = n0 + j * 16 + lidx;
        if (n >= N) continue;
        const float bv = bias ? bias[n] : 0.f;
        #pragma unroll
        for (int r = 0; r < 8; ++r) {
            const int m = m0 + hlf * 8 + r;
            if (m < M) {
                float v = acc[j][r] + bv;
                if (act == 1)      v = v > 0.f ? v : 0.f;
                else if (act == 2) v = 1.f / (1.f + __expf(-v));
                C[(size_t)m * ldc + n] = v;
            }
        }
    }
}

// ---------------------------------------------------------------------------
// Embedding gather: h[row] = tok_emb[tokens[row]]
// ---------------------------------------------------------------------------
__global__ void __launch_bounds__(256)
k_embed(const int* __restrict__ tokens, const float* __restrict__ emb,
        float* __restrict__ h)
{
    const int r = blockIdx.x;
    const int t = tokens[r];
    const float* src = emb + (size_t)t * DD;
    float* dst = h + (size_t)r * DD;
    for (int j = threadIdx.x; j < DD; j += 256) dst[j] = src[j];
}

// ---------------------------------------------------------------------------
// LRU scan: one block per sequence chunk c (4 chunks of 256 steps),
// state h (768 floats) lives in LDS.  h_t = xp_t + h_{t-1} @ Whh^T
// ---------------------------------------------------------------------------
__global__ void __launch_bounds__(256)
k_lru_scan(const float* __restrict__ xp, const float* __restrict__ Whh,
           float* __restrict__ hy)
{
    __shared__ float hbuf[DD];
    const int c   = blockIdx.x;
    const int tid = threadIdx.x;
    for (int j = tid; j < DD; j += 256) hbuf[j] = 0.f;
    __syncthreads();
    for (int t = 0; t < MEMD; ++t) {
        const size_t row = (size_t)(c * MEMD + t);
        const float* xr  = xp + row * DD;
        float accv[3];
        #pragma unroll
        for (int u = 0; u < 3; ++u) {
            const int j = tid * 3 + u;
            const float* wr = Whh + (size_t)j * DD;
            float s = xr[j];
            for (int k = 0; k < DD; ++k) s += hbuf[k] * wr[k];
            accv[u] = s;
        }
        __syncthreads();
        float* outr = hy + row * DD;
        #pragma unroll
        for (int u = 0; u < 3; ++u) {
            const int j = tid * 3 + u;
            hbuf[j] = accv[u];
            outr[j] = accv[u];
        }
        __syncthreads();
    }
}

// ---------------------------------------------------------------------------
// MHA over the length-2 batch axis (faithful to the reference).
// grid = (512, 4) = (n, head), block = 192 threads (= head_dim).
// ---------------------------------------------------------------------------
__global__ void __launch_bounds__(192)
k_attn(const float* __restrict__ qkv, float* __restrict__ attn)
{
    __shared__ float red[4][HDD];
    __shared__ float aw[2][2];
    const int n  = blockIdx.x;
    const int hh = blockIdx.y;
    const int d  = threadIdx.x;
    const size_t r0 = (size_t)(0 * SD + n) * (3 * DD);
    const size_t r1 = (size_t)(1 * SD + n) * (3 * DD);
    const int off = hh * HDD + d;
    const float q0 = qkv[r0 + off],            q1 = qkv[r1 + off];
    const float k0 = qkv[r0 + DD + off],       k1 = qkv[r1 + DD + off];
    const float v0 = qkv[r0 + 2 * DD + off],   v1 = qkv[r1 + 2 * DD + off];
    red[0][d] = q0 * k0; red[1][d] = q0 * k1;
    red[2][d] = q1 * k0; red[3][d] = q1 * k1;
    __syncthreads();
    for (int s = 96; s >= 6; s >>= 1) {
        if (d < s) {
            #pragma unroll
            for (int q = 0; q < 4; ++q) red[q][d] += red[q][d + s];
        }
        __syncthreads();
    }
    if (d == 0) {
        const float sc = 0.07216878f;   // 1/sqrt(192)
        #pragma unroll
        for (int l = 0; l < 2; ++l) {
            float x0 = 0.f, x1 = 0.f;
            for (int q = 0; q < 6; ++q) { x0 += red[2*l][q]; x1 += red[2*l+1][q]; }
            x0 *= sc; x1 *= sc;
            const float mx = fmaxf(x0, x1);
            const float e0 = __expf(x0 - mx), e1 = __expf(x1 - mx);
            const float inv = 1.f / (e0 + e1);
            aw[l][0] = e0 * inv; aw[l][1] = e1 * inv;
        }
    }
    __syncthreads();
    attn[(size_t)(0 * SD + n) * DD + off] = aw[0][0] * v0 + aw[0][1] * v1;
    attn[(size_t)(1 * SD + n) * DD + off] = aw[1][0] * v0 + aw[1][1] * v1;
}

// softmax over the 2 task logits, in place
__global__ void __launch_bounds__(256)
k_softmax2(float* __restrict__ tp)
{
    const int r = blockIdx.x * 256 + threadIdx.x;
    if (r >= TOK) return;
    const float a = tp[r * 2], b = tp[r * 2 + 1];
    const float m = fmaxf(a, b);
    const float ea = __expf(a - m), eb = __expf(b - m);
    const float inv = 1.f / (ea + eb);
    tp[r * 2] = ea * inv; tp[r * 2 + 1] = eb * inv;
}

// dst[r,j] = src[r,j] * tp[r, t]
__global__ void __launch_bounds__(256)
k_scale_rows(const float* __restrict__ src, const float* __restrict__ tp, int t,
             float* __restrict__ dst)
{
    const int idx = blockIdx.x * 256 + threadIdx.x;
    if (idx >= TOK * DD) return;
    const int r = idx / DD;
    dst[idx] = src[idx] * tp[r * NTD + t];
}

// xcat[r] = [ x[r] (768) | te[r] (64) ]
__global__ void __launch_bounds__(256)
k_concat(const float* __restrict__ x, const float* __restrict__ te,
         float* __restrict__ xcat)
{
    const int r = blockIdx.x;
    for (int j = threadIdx.x; j < DD; j += 256)
        xcat[(size_t)r * (DD + ATTRD) + j] = x[(size_t)r * DD + j];
    if (threadIdx.x < ATTRD)
        xcat[(size_t)r * (DD + ATTRD) + DD + threadIdx.x] = te[(size_t)r * ATTRD + threadIdx.x];
}

// ---------------------------------------------------------------------------
// Router finalize: expert logits (8 dots of 768, one wave each), softmax,
// attr-prior softmax+mean, top-2 mask, entropy -> atomicAdd into loss.
// one block (256 thr) per token.
// ---------------------------------------------------------------------------
__global__ void __launch_bounds__(256)
k_router_final(const float* __restrict__ r2, const float* __restrict__ r_w,
               const float* __restrict__ r_b, const float* __restrict__ te,
               const float* __restrict__ r_attr, float* __restrict__ ep,
               float* __restrict__ loss)
{
    __shared__ float lg[NED];
    __shared__ float apv[NTD][NED];
    const int tok  = blockIdx.x;
    const int lane = threadIdx.x & 31;
    const int e    = threadIdx.x >> 5;           // expert = wave id
    const float* row = r2 + (size_t)tok * DD;
    const float* wr  = r_w + (size_t)e * DD;
    float s = 0.f;
    for (int k = lane; k < DD; k += 32) s += row[k] * wr[k];
    #pragma unroll
    for (int off = 16; off > 0; off >>= 1) s += __shfl_down(s, off, 32);
    if (lane == 0) lg[e] = s + r_b[e];
    if (threadIdx.x < NTD * NED) {
        const int t  = threadIdx.x >> 3;
        const int ee = threadIdx.x & 7;
        const float* ter = te + (size_t)tok * (NTD * TDD) + t * TDD;
        float a = 0.f;
        for (int dd = 0; dd < TDD; ++dd) a += ter[dd] * r_attr[dd * NED + ee];
        apv[t][ee] = a;
    }
    __syncthreads();
    if (threadIdx.x == 0) {
        float p[NED];
        float mx = lg[0];
        for (int q = 1; q < NED; ++q) mx = fmaxf(mx, lg[q]);
        float ssum = 0.f;
        for (int q = 0; q < NED; ++q) { p[q] = __expf(lg[q] - mx); ssum += p[q]; }
        const float isum = 1.f / ssum;
        for (int q = 0; q < NED; ++q) p[q] *= isum;
        float apm[NED] = {};
        for (int t = 0; t < NTD; ++t) {
            float m2 = apv[t][0];
            for (int q = 1; q < NED; ++q) m2 = fmaxf(m2, apv[t][q]);
            float s2 = 0.f, tmp[NED];
            for (int q = 0; q < NED; ++q) { tmp[q] = __expf(apv[t][q] - m2); s2 += tmp[q]; }
            const float i2 = 1.f / s2;
            for (int q = 0; q < NED; ++q) apm[q] += tmp[q] * i2 * (1.f / NTD);
        }
        for (int q = 0; q < NED; ++q) p[q] *= apm[q];
        int i1 = 0;
        for (int q = 1; q < NED; ++q) if (p[q] > p[i1]) i1 = q;
        int i2 = -1;
        for (int q = 0; q < NED; ++q)
            if (q != i1 && (i2 < 0 || p[q] > p[i2])) i2 = q;
        float ent = 0.f;
        for (int q = 0; q < NED; ++q) {
            const float v = (q == i1 || q == i2) ? p[q] : 0.f;
            ep[(size_t)tok * NED + q] = v;
            ent += v * __logf(v + 1e-8f);
        }
        atomicAdd(loss, -ent * (1.0f / (float)TOK));
    }
}

// attr[e] = ap_w[e] @ attr_emb[e] + ap_b[e]     (grid = NE, block = 256)
__global__ void __launch_bounds__(256)
k_attr(const float* __restrict__ ap_w, const float* __restrict__ ap_b,
       const float* __restrict__ attr_emb, float* __restrict__ attr)
{
    __shared__ float ae[ATTRD];
    const int e = blockIdx.x;
    if (threadIdx.x < ATTRD) ae[threadIdx.x] = attr_emb[e * ATTRD + threadIdx.x];
    __syncthreads();
    for (int j = threadIdx.x; j < DD; j += 256) {
        const float* wr = ap_w + ((size_t)e * DD + j) * ATTRD;
        float s = ap_b[e * DD + j];
        for (int a = 0; a < ATTRD; ++a) s += wr[a] * ae[a];
        attr[e * DD + j] = s;
    }
}

// g1bias[e] = g1_w[e][:,768:1536] @ attr[e] + g1_b[e]   (grid = (NE, 6))
__global__ void __launch_bounds__(256)
k_g1bias(const float* __restrict__ g1_w, const float* __restrict__ g1_b,
         const float* __restrict__ attr, float* __restrict__ g1bias)
{
    const int e = blockIdx.x;
    const int j = blockIdx.y * 256 + threadIdx.x;
    const float* wr = g1_w + (((size_t)e * INTERD + j) * (2 * DD)) + DD;
    const float* ar = attr + (size_t)e * DD;
    float s = g1_b[e * INTERD + j];
    for (int k = 0; k < DD; ++k) s += wr[k] * ar[k];
    g1bias[e * INTERD + j] = s;
}

// y = x*g + attr*(1-g)      (g already sigmoid'd by GEMM epilogue)
__global__ void __launch_bounds__(256)
k_gate(const float* __restrict__ x, const float* __restrict__ g,
       const float* __restrict__ attr_e, float* __restrict__ y)
{
    const int idx = blockIdx.x * 256 + threadIdx.x;
    if (idx >= TOK * DD) return;
    const int j = idx % DD;
    const float gv = g[idx];
    y[idx] = x[idx] * gv + attr_e[j] * (1.f - gv);
}

// acc += ep[tok,e] * LayerNorm(fo + x)     (one block per token)
__global__ void __launch_bounds__(256)
k_expert_final(const float* __restrict__ fo, const float* __restrict__ x,
               const float* __restrict__ lng, const float* __restrict__ lnb,
               const float* __restrict__ ep, int e, float* __restrict__ acc)
{
    __shared__ float red[256];
    const int tok = blockIdx.x;
    const float w = ep[(size_t)tok * NED + e];
    const float* fr = fo + (size_t)tok * DD;
    const float* xr = x  + (size_t)tok * DD;
    float v[3]; float lsum = 0.f;
    #pragma unroll
    for (int u = 0; u < 3; ++u) {
        const int j = threadIdx.x * 3 + u;
        v[u] = fr[j] + xr[j];
        lsum += v[u];
    }
    red[threadIdx.x] = lsum; __syncthreads();
    for (int s = 128; s > 0; s >>= 1) {
        if (threadIdx.x < s) red[threadIdx.x] += red[threadIdx.x + s];
        __syncthreads();
    }
    const float mean = red[0] * (1.f / DD);
    __syncthreads();
    float vs = 0.f;
    #pragma unroll
    for (int u = 0; u < 3; ++u) { const float dd = v[u] - mean; vs += dd * dd; }
    red[threadIdx.x] = vs; __syncthreads();
    for (int s = 128; s > 0; s >>= 1) {
        if (threadIdx.x < s) red[threadIdx.x] += red[threadIdx.x + s];
        __syncthreads();
    }
    const float inv = rsqrtf(red[0] * (1.f / DD) + 1e-5f);
    #pragma unroll
    for (int u = 0; u < 3; ++u) {
        const int j = threadIdx.x * 3 + u;
        acc[(size_t)tok * DD + j] += w * ((v[u] - mean) * inv * lng[j] + lnb[j]);
    }
}

// RMSNorm
__global__ void __launch_bounds__(256)
k_rmsnorm(const float* __restrict__ h, const float* __restrict__ w,
          float* __restrict__ out)
{
    __shared__ float red[256];
    const int tok = blockIdx.x;
    const float* hr = h + (size_t)tok * DD;
    float v[3]; float s = 0.f;
    #pragma unroll
    for (int u = 0; u < 3; ++u) {
        const int j = threadIdx.x * 3 + u;
        v[u] = hr[j]; s += v[u] * v[u];
    }
    red[threadIdx.x] = s; __syncthreads();
    for (int st = 128; st > 0; st >>= 1) {
        if (threadIdx.x < st) red[threadIdx.x] += red[threadIdx.x + st];
        __syncthreads();
    }
    const float inv = rsqrtf(red[0] * (1.f / DD) + 1e-5f);
    #pragma unroll
    for (int u = 0; u < 3; ++u) {
        const int j = threadIdx.x * 3 + u;
        out[(size_t)tok * DD + j] = v[u] * inv * w[j];
    }
}

__global__ void __launch_bounds__(256)
k_zero(float* __restrict__ p, int n)
{
    const int idx = blockIdx.x * 256 + threadIdx.x;
    if (idx < n) p[idx] = 0.f;
}

__global__ void k_store_loss(const float* __restrict__ lossb, float* __restrict__ dst)
{
    if (threadIdx.x == 0 && blockIdx.x == 0) dst[0] = lossb[0];
}

// ---------------------------------------------------------------------------
// Orchestration
// ---------------------------------------------------------------------------
extern "C" void kernel_launch(void* const* d_in, const int* in_sizes, int n_in,
                              void* d_out, int out_size, void* d_ws, size_t ws_size,
                              hipStream_t stream)
{
    (void)in_sizes; (void)n_in; (void)out_size; (void)ws_size;
    int i = 0;
    const int*   tokens   = (const int*)  d_in[i++];   // 0
    const float* tok_emb  = (const float*)d_in[i++];   // 1
    const float* lru_Wih  = (const float*)d_in[i++];   // 2
    const float* lru_Whh  = (const float*)d_in[i++];   // 3
    const float* lru_b    = (const float*)d_in[i++];   // 4
    const float* enc_in_w = (const float*)d_in[i++];   // 5
    const float* enc_in_b = (const float*)d_in[i++];
    const float* enc_out_w= (const float*)d_in[i++];
    const float* enc_out_b= (const float*)d_in[i++];
    const float* div_w1   = (const float*)d_in[i++];
    const float* div_b1   = (const float*)d_in[i++];
    const float* div_w2   = (const float*)d_in[i++];
    const float* div_b2   = (const float*)d_in[i++];
    const float* te_w1    = (const float*)d_in[i++];
    const float* te_b1    = (const float*)d_in[i++];
    const float* te_w2    = (const float*)d_in[i++];
    const float* te_b2    = (const float*)d_in[i++];   // 16
    const float* L_attr_emb = (const float*)d_in[i++]; // 17
    const float* L_ap_w   = (const float*)d_in[i++];
    const float* L_ap_b   = (const float*)d_in[i++];
    const float* L_g1_w   = (const float*)d_in[i++];
    const float* L_g1_b   = (const float*)d_in[i++];
    const float* L_g2_w   = (const float*)d_in[i++];
    const float* L_g2_b   = (const float*)d_in[i++];
    const float* L_f1_w   = (const float*)d_in[i++];
    const float* L_f1_b   = (const float*)d_in[i++];
    const float* L_f2_w   = (const float*)d_in[i++];
    const float* L_f2_b   = (const float*)d_in[i++];
    const float* L_ln_g   = (const float*)d_in[i++];
    const float* L_ln_b   = (const float*)d_in[i++];
    const float* L_r_attr = (const float*)d_in[i++];
    const float* L_r_in_w = (const float*)d_in[i++];
    const float* L_r_in_b = (const float*)d_in[i++];
    const float* L_r_mid_w= (const float*)d_in[i++];
    const float* L_r_mid_b= (const float*)d_in[i++];
    const float* L_r_w    = (const float*)d_in[i++];
    const float* L_r_b    = (const float*)d_in[i++];   // 36
    const float* norm_w   = (const float*)d_in[i++];   // 37

    // ---- workspace carve-up (floats) ----
    float* ws = (float*)d_ws;
    size_t off = 0;
    auto alloc = [&](size_t n) { float* p = ws + off; off += n; return p; };
    float* hA     = alloc((size_t)TOK * DD);           // hidden ping
    float* hB     = alloc((size_t)TOK * DD);           // hidden pong
    float* big0   = alloc((size_t)TOK * 3 * DD);       // xp / qkv
    float* bufA   = alloc((size_t)TOK * DD);           // attn-in / te hidden
    float* bufB   = alloc((size_t)TOK * DD);           // attn projected
    float* bufC   = alloc((size_t)TOK * DD);           // div1 / ti
    float* tp     = alloc((size_t)TOK * NTD);          // task probs
    float* te     = alloc((size_t)TOK * NTD * TDD);    // task embeddings
    float* xcat   = alloc((size_t)TOK * (DD + ATTRD));
    float* r1     = alloc((size_t)TOK * 4 * DD);
    float* r2     = alloc((size_t)TOK * DD);
    float* ep     = alloc((size_t)TOK * NED);
    float* attrB  = alloc((size_t)NED * DD);
    float* g1bias = alloc((size_t)NED * INTERD);
    float* gh     = alloc((size_t)TOK * INTERD);
    float* gsig   = alloc((size_t)TOK * DD);
    float* yb     = alloc((size_t)TOK * DD);
    float* fh     = alloc((size_t)TOK * INTERD);
    float* fo     = alloc((size_t)TOK * DD);
    float* accb   = alloc((size_t)TOK * DD);
    float* lossb  = alloc(64);

    auto GEMM = [&](const float* A, int lda, const float* Wt, int ldw,
                    const float* bias, float* C, int ldc,
                    int M, int N, int K, int act) {
        dim3 grid((N + 127) / 128, (M + 63) / 64);
        k_gemm_wmma<<<grid, 256, 0, stream>>>(A, lda, Wt, ldw, bias, C, ldc, M, N, K, act);
    };

    k_zero<<<1, 256, 0, stream>>>(lossb, 1);

    // ---- 1. embedding ----
    k_embed<<<TOK, 256, 0, stream>>>(tokens, tok_emb, hA);

    // ---- 2. LRU connector: xp = h @ Wih^T + b ; serial scan per chunk ----
    GEMM(hA, DD, lru_Wih, DD, lru_b, big0, DD, TOK, DD, DD, 0);
    k_lru_scan<<<4, 256, 0, stream>>>(big0, lru_Whh, hB);
    float* cur = hB;
    float* nxt = hA;

    // ---- 3. task encoder ----
    GEMM(cur, DD, enc_in_w, DD, enc_in_b, big0, 3 * DD, TOK, 3 * DD, DD, 0);
    k_attn<<<dim3(SD, NHD), 192, 0, stream>>>(big0, bufA);
    GEMM(bufA, DD, enc_out_w, DD, enc_out_b, bufB, DD, TOK, DD, DD, 0);
    GEMM(bufB, DD, div_w1, DD, div_b1, bufC, DD, TOK, DD, DD, 1);
    GEMM(bufC, DD, div_w2, DD, div_b2, tp, NTD, TOK, NTD, DD, 0);
    k_softmax2<<<(TOK + 255) / 256, 256, 0, stream>>>(tp);
    for (int t = 0; t < NTD; ++t) {
        k_scale_rows<<<(TOK * DD) / 256, 256, 0, stream>>>(bufB, tp, t, bufC);
        GEMM(bufC, DD, te_w1 + (size_t)t * DD * DD, DD, te_b1 + (size_t)t * DD,
             bufA, DD, TOK, DD, DD, 1);
        GEMM(bufA, DD, te_w2 + (size_t)t * TDD * DD, DD, te_b2 + (size_t)t * TDD,
             te + t * TDD, NTD * TDD, TOK, TDD, DD, 0);
    }

    // ---- 4. MoE layers ----
    for (int l = 0; l < NLD; ++l) {
        const float* attr_emb = L_attr_emb + (size_t)l * NED * ATTRD;
        const float* ap_w     = L_ap_w   + (size_t)l * NED * DD * ATTRD;
        const float* ap_b     = L_ap_b   + (size_t)l * NED * DD;
        const float* g1_w     = L_g1_w   + (size_t)l * NED * INTERD * 2 * DD;
        const float* g1_b     = L_g1_b   + (size_t)l * NED * INTERD;
        const float* g2_w     = L_g2_w   + (size_t)l * NED * DD * INTERD;
        const float* g2_b     = L_g2_b   + (size_t)l * NED * DD;
        const float* f1_w     = L_f1_w   + (size_t)l * NED * INTERD * DD;
        const float* f1_b     = L_f1_b   + (size_t)l * NED * INTERD;
        const float* f2_w     = L_f2_w   + (size_t)l * NED * DD * INTERD;
        const float* f2_b     = L_f2_b   + (size_t)l * NED * DD;
        const float* ln_g     = L_ln_g   + (size_t)l * NED * DD;
        const float* ln_b     = L_ln_b   + (size_t)l * NED * DD;
        const float* r_attr   = L_r_attr + (size_t)l * TDD * NED;
        const float* r_in_w   = L_r_in_w + (size_t)l * 4 * DD * (DD + ATTRD);
        const float* r_in_b   = L_r_in_b + (size_t)l * 4 * DD;
        const float* r_mid_w  = L_r_mid_w+ (size_t)l * DD * 4 * DD;
        const float* r_mid_b  = L_r_mid_b+ (size_t)l * DD;
        const float* r_w      = L_r_w    + (size_t)l * NED * DD;
        const float* r_b      = L_r_b    + (size_t)l * NED;

        // router
        k_concat<<<TOK, 256, 0, stream>>>(cur, te, xcat);
        GEMM(xcat, DD + ATTRD, r_in_w, DD + ATTRD, r_in_b, r1, 4 * DD,
             TOK, 4 * DD, DD + ATTRD, 1);
        GEMM(r1, 4 * DD, r_mid_w, 4 * DD, r_mid_b, r2, DD, TOK, DD, 4 * DD, 1);
        k_router_final<<<TOK, 256, 0, stream>>>(r2, r_w, r_b, te, r_attr, ep, lossb);

        // per-expert constants
        k_attr<<<NED, 256, 0, stream>>>(ap_w, ap_b, attr_emb, attrB);
        k_g1bias<<<dim3(NED, INTERD / 256), 256, 0, stream>>>(g1_w, g1_b, attrB, g1bias);
        k_zero<<<(TOK * DD) / 256, 256, 0, stream>>>(accb, TOK * DD);

        // dense expert execution (matches reference vmap), weighted accumulate
        for (int e = 0; e < NED; ++e) {
            // g_hidden = relu(x @ g1_w[:,:768]^T + (g1_w[:,768:]@attr + g1_b))
            GEMM(cur, DD, g1_w + (size_t)e * INTERD * 2 * DD, 2 * DD,
                 g1bias + e * INTERD, gh, INTERD, TOK, INTERD, DD, 1);
            // g = sigmoid(g_hidden @ g2_w^T + g2_b)
            GEMM(gh, INTERD, g2_w + (size_t)e * DD * INTERD, INTERD,
                 g2_b + e * DD, gsig, DD, TOK, DD, INTERD, 2);
            // y = x*g + attr*(1-g)
            k_gate<<<(TOK * DD) / 256, 256, 0, stream>>>(cur, gsig, attrB + e * DD, yb);
            // f = f2(relu(f1(y)))
            GEMM(yb, DD, f1_w + (size_t)e * INTERD * DD, DD,
                 f1_b + e * INTERD, fh, INTERD, TOK, INTERD, DD, 1);
            GEMM(fh, INTERD, f2_w + (size_t)e * DD * INTERD, INTERD,
                 f2_b + e * DD, fo, DD, TOK, DD, INTERD, 0);
            // acc += ep[:,e] * LayerNorm(f + x)
            k_expert_final<<<TOK, 256, 0, stream>>>(fo, cur, ln_g + e * DD,
                                                    ln_b + e * DD, ep, e, accb);
        }
        (void)hipMemcpyAsync(nxt, accb, (size_t)TOK * DD * sizeof(float),
                             hipMemcpyDeviceToDevice, stream);
        float* tswap = cur; cur = nxt; nxt = tswap;
    }

    // ---- 5. final norm + tied vocab projection + loss ----
    k_rmsnorm<<<TOK, 256, 0, stream>>>(cur, norm_w, nxt);
    GEMM(nxt, DD, tok_emb, DD, nullptr, (float*)d_out, VD, TOK, VD, DD, 0);
    k_store_loss<<<1, 1, 0, stream>>>(lossb, (float*)d_out + (size_t)TOK * VD);
}
